// TemporalConsistencySSM_3882650437059
// MI455X (gfx1250) — matrix operand order repaired and checked by reading, not compiled, not run on hardware
//
#include <hip/hip_runtime.h>
#include <hip/hip_bf16.h>
#include <math.h>

// ---------------------------------------------------------------------------
// TemporalConsistencySSM (Mamba block) for gfx1250 / MI455X, wave32 + WMMA.
// B=2, L=1024, D_MODEL=512, D_INNER=1024, D_STATE=64, DT_RANK=32, D_CONV=4
// ---------------------------------------------------------------------------

#define D_MODEL 512
#define D_STATE 64
#define D_INNER 1024
#define D_CONV  4
#define DT_RANK 32
#define NBATCH  2
#define SEQLEN  1024
#define NTOK    (NBATCH * SEQLEN)          // 2048 rows
#define XDB_W   (DT_RANK + 2 * D_STATE)    // 160 (logical)
#define XDB_P   192                        // padded to a multiple of 64
#define LN_EPS  1e-5f

typedef __attribute__((ext_vector_type(16))) _Float16 v16h;
typedef __attribute__((ext_vector_type(8)))  _Float16 v8h;
typedef __attribute__((ext_vector_type(8)))  float    v8f;
typedef __attribute__((ext_vector_type(4)))  int      v4i;

// ---------------------------------------------------------------------------
// Async global -> LDS copy (ASYNCcnt path, CDNA5 ISA).
// Builtin takes AS1 int4* (global) and AS3 int4* (LDS) per hipcc diagnostic.
// ---------------------------------------------------------------------------
#if __has_builtin(__builtin_amdgcn_global_load_async_to_lds_b128) && \
    __has_builtin(__builtin_amdgcn_s_wait_asynccnt)
#define HAVE_ASYNC_LDS 1
typedef __attribute__((address_space(1))) v4i gv4i;
typedef __attribute__((address_space(3))) v4i lv4i;
#endif

__device__ __forceinline__ void async_copy_b128(const _Float16* g, _Float16* l) {
#ifdef HAVE_ASYNC_LDS
    __builtin_amdgcn_global_load_async_to_lds_b128((gv4i*)g, (lv4i*)l, 0, 0);
#else
    *(v8h*)l = *(const v8h*)g;
#endif
}

__device__ __forceinline__ void async_wait0() {
#ifdef HAVE_ASYNC_LDS
    __builtin_amdgcn_s_wait_asynccnt(0);
#endif
}

// ---------------------------------------------------------------------------
// Weight transpose + f32 -> f16 convert:  in (R x C) row-major -> out (C x R)
// ---------------------------------------------------------------------------
__global__ void k_transpose_cvt(const float* __restrict__ in,
                                _Float16* __restrict__ out, int R, int C) {
    int idx = blockIdx.x * blockDim.x + threadIdx.x;
    if (idx >= R * C) return;
    int r = idx / C, c = idx - r * C;
    out[(size_t)c * R + r] = (_Float16)in[idx];
}

// ---------------------------------------------------------------------------
// LayerNorm over last dim (512), emit f16 activations for the WMMA GEMM.
// ---------------------------------------------------------------------------
__global__ void k_layernorm(const float* __restrict__ frames,
                            const float* __restrict__ gamma,
                            const float* __restrict__ beta,
                            _Float16* __restrict__ xn16) {
    __shared__ float red[256];
    const int row = blockIdx.x;
    const float* in = frames + (size_t)row * D_MODEL;

    float s = 0.f;
    for (int i = threadIdx.x; i < D_MODEL; i += 256) s += in[i];
    red[threadIdx.x] = s;
    __syncthreads();
    for (int off = 128; off > 0; off >>= 1) {
        if (threadIdx.x < off) red[threadIdx.x] += red[threadIdx.x + off];
        __syncthreads();
    }
    const float mu = red[0] * (1.f / D_MODEL);
    __syncthreads();

    float v = 0.f;
    for (int i = threadIdx.x; i < D_MODEL; i += 256) {
        float d = in[i] - mu;
        v += d * d;
    }
    red[threadIdx.x] = v;
    __syncthreads();
    for (int off = 128; off > 0; off >>= 1) {
        if (threadIdx.x < off) red[threadIdx.x] += red[threadIdx.x + off];
        __syncthreads();
    }
    const float rs = rsqrtf(red[0] * (1.f / D_MODEL) + LN_EPS);

    for (int i = threadIdx.x; i < D_MODEL; i += 256) {
        float y = (in[i] - mu) * rs * gamma[i] + beta[i];
        xn16[(size_t)row * D_MODEL + i] = (_Float16)y;
    }
}

// ---------------------------------------------------------------------------
// WMMA GEMM:  C(MxN,f32) = A(MxK,f16 rowmaj) * B with B given transposed as
// BT (NxK, f16 rowmaj).  4-wave workgroup computes a 64x64 macro-tile:
//   - B panel (64 cols x 32 k) staged into LDS with async global->LDS copies,
//     double-buffered; shared by all 4 waves.
//   - each wave owns 16 rows, reuses its A fragment across 4 accumulators.
// Fragment layouts follow CDNA5 ISA.
// EPI: 0 = plain store, 1 = softplus(acc + bias[n]), 2 = acc + resid[m,n]
// ---------------------------------------------------------------------------
#define GEMM_BM 64
#define GEMM_BN 64
#define LDSB_STRIDE 40   // halves per staged row (32 data + 8 pad)

template <int EPI>
__global__ __launch_bounds__(128)
void k_wmma_gemm(const _Float16* __restrict__ A,
                 const _Float16* __restrict__ BT,
                 float* __restrict__ Cout,
                 int M, int N, int K,
                 const float* __restrict__ bias,
                 const float* __restrict__ resid) {
    __shared__ _Float16 smB[2][GEMM_BN * LDSB_STRIDE];

    const int nBase = blockIdx.x * GEMM_BN;
    const int mBase = blockIdx.y * GEMM_BM;
    const int tid   = threadIdx.x;
    const int wave  = tid >> 5;
    const int lane  = tid & 31;
    const int half  = lane >> 4;          // 0 | 1
    const int l16   = lane & 15;

    const int mRow = mBase + wave * 16 + l16;
    const _Float16* aRow = A + (size_t)mRow * K + half * 8;

    // Stage BT[nBase..nBase+63][k..k+31] -> smB[buf]: 256 x 16B chunks,
    // 2 chunks per thread, each an async b128 global->LDS transfer.
    auto stage = [&](int buf, int k) {
        #pragma unroll
        for (int c = 0; c < 2; ++c) {
            const int chunk = tid * 2 + c;          // 0..255
            const int row   = chunk >> 2;           // 0..63
            const int off   = chunk & 3;            // 16B chunk within row
            const _Float16* g = BT + (size_t)(nBase + row) * K + k + off * 8;
            _Float16* l = &smB[buf][row * LDSB_STRIDE + off * 8];
            async_copy_b128(g, l);
        }
    };

    v8f acc[4] = {};
    stage(0, 0);
    int cur = 0;

    for (int k = 0; k < K; k += 32) {
        async_wait0();          // this wave's panel chunks landed
        __syncthreads();        // every wave's chunks landed; prior reads done
        if (k + 32 < K) stage(cur ^ 1, k + 32);

        // A frag (direct from global, contiguous 16B per lane):
        //   lanes 0-15: elems 0-7 = K[k..k+7],  8-15 = K[k+16..k+23]
        //   lanes16-31: elems 0-7 = K[k+8..],   8-15 = K[k+24..]
        if (k + 64 < K) __builtin_prefetch((const void*)(aRow + k + 64), 0, 1);
        const v8h a0 = *(const v8h*)(aRow + k);
        const v8h a1 = *(const v8h*)(aRow + k + 16);
        union { v16h v; v8h h[2]; } au;
        au.h[0] = a0;
        au.h[1] = a1;

        #pragma unroll
        for (int n = 0; n < 4; ++n) {
            // B frag from LDS: lanes 0-15 -> N=l16, K[k..k+7],K[k+16..k+23]; 16-31 -> K[k+8..],K[k+24..]
            const _Float16* bp = &smB[cur][(n * 16 + l16) * LDSB_STRIDE + half * 8];
            const v8h b0 = *(const v8h*)bp;
            const v8h b1 = *(const v8h*)(bp + 16);
            union { v16h v; v8h h[2]; } bu;
            bu.h[0] = b0;
            bu.h[1] = b1;
            acc[n] = __builtin_amdgcn_wmma_f32_16x16x32_f16(
                false, au.v, false, bu.v, (short)0, acc[n], false, false);
        }
        cur ^= 1;
    }

    // D layout: VGPR r -> M = half*8 + r (within tile), N = l16
    #pragma unroll
    for (int n = 0; n < 4; ++n) {
        const int ncol = nBase + n * 16 + l16;
        #pragma unroll
        for (int r = 0; r < 8; ++r) {
            const int m = mBase + wave * 16 + half * 8 + r;
            float v = acc[n][r];
            if (EPI == 1) {
                v += bias[ncol];
                v = (v > 20.f) ? v : log1pf(__expf(v));   // softplus
            } else if (EPI == 2) {
                v += resid[(size_t)m * N + ncol];
            }
            Cout[(size_t)m * N + ncol] = v;
        }
    }
}

// ---------------------------------------------------------------------------
// Causal depthwise conv1d (taps=4) along time + bias + SiLU.
// ---------------------------------------------------------------------------
__global__ void k_conv_silu(const float* __restrict__ xz,
                            const float* __restrict__ conv_w,
                            const float* __restrict__ conv_b,
                            float* __restrict__ xf32,
                            _Float16* __restrict__ xf16) {
    int idx = blockIdx.x * blockDim.x + threadIdx.x;   // over NTOK * D_INNER
    if (idx >= NTOK * D_INNER) return;
    const int d = idx & (D_INNER - 1);
    const int t = (idx >> 10) & (SEQLEN - 1);
    const int b = idx >> 20;
    const float* col = xz + (size_t)b * SEQLEN * (2 * D_INNER) + d;

    float acc = conv_b[d];
    #pragma unroll
    for (int j = 0; j < D_CONV; ++j) {
        int ts = t - (D_CONV - 1) + j;
        if (ts >= 0) acc += conv_w[d * D_CONV + j] * col[(size_t)ts * (2 * D_INNER)];
    }
    float y = acc / (1.f + __expf(-acc));              // SiLU
    xf32[idx] = y;
    xf16[idx] = (_Float16)y;
}

// dt slice of xdb -> f16 A-matrix for the delta GEMM (2048 x 32)
__global__ void k_extract_dt(const float* __restrict__ xdb,
                             _Float16* __restrict__ dt16) {
    int idx = blockIdx.x * blockDim.x + threadIdx.x;
    if (idx >= NTOK * DT_RANK) return;
    int row = idx >> 5, j = idx & (DT_RANK - 1);
    dt16[idx] = (_Float16)xdb[(size_t)row * XDB_P + j];
}

// ---------------------------------------------------------------------------
// Selective scan.  One wave32 per (batch, channel); each lane owns states
// s = lane and s = lane + 32.  Sequential over t, wave-reduce y.
// ---------------------------------------------------------------------------
__global__ void k_scan(const float* __restrict__ delta,   // (NTOK, D_INNER)
                       const float* __restrict__ xdb,     // (NTOK, XDB_P)
                       const float* __restrict__ xf32,    // (NTOK, D_INNER)
                       const float* __restrict__ A_log,   // (D_INNER, D_STATE)
                       float* __restrict__ ys) {          // (NTOK, D_INNER)
    const int bd   = blockIdx.x;           // 0..2047
    const int b    = bd >> 10;
    const int d    = bd & (D_INNER - 1);
    const int lane = threadIdx.x;          // 0..31

    const float A0 = -__expf(A_log[(size_t)d * D_STATE + lane]);
    const float A1 = -__expf(A_log[(size_t)d * D_STATE + lane + 32]);

    float h0 = 0.f, h1 = 0.f;
    const size_t row0 = (size_t)b * SEQLEN;
    for (int t = 0; t < SEQLEN; ++t) {
        const size_t row = row0 + t;
        const float dt = delta[row * D_INNER + d];
        const float xt = xf32[row * D_INNER + d];
        const float* bc = xdb + row * XDB_P;
        const float B0 = bc[DT_RANK + lane];
        const float B1 = bc[DT_RANK + lane + 32];
        const float C0 = bc[DT_RANK + D_STATE + lane];
        const float C1 = bc[DT_RANK + D_STATE + lane + 32];

        const float du = dt * xt;
        h0 = h0 * __expf(dt * A0) + du * B0;
        h1 = h1 * __expf(dt * A1) + du * B1;

        float p = h0 * C0 + h1 * C1;
        #pragma unroll
        for (int off = 16; off > 0; off >>= 1)
            p += __shfl_xor(p, off, 32);
        if (lane == 0) ys[row * D_INNER + d] = p;
    }
}

// y = (ys + x*D) * silu(z) -> f16 for out-projection GEMM
__global__ void k_gate(const float* __restrict__ ys,
                       const float* __restrict__ xf32,
                       const float* __restrict__ xz,
                       const float* __restrict__ Dvec,
                       _Float16* __restrict__ y16) {
    int idx = blockIdx.x * blockDim.x + threadIdx.x;
    if (idx >= NTOK * D_INNER) return;
    const int d   = idx & (D_INNER - 1);
    const int row = idx >> 10;
    const float z = xz[(size_t)row * (2 * D_INNER) + D_INNER + d];
    const float y = (ys[idx] + xf32[idx] * Dvec[d]) * (z / (1.f + __expf(-z)));
    y16[idx] = (_Float16)y;
}

// ---------------------------------------------------------------------------
// Launch
// ---------------------------------------------------------------------------
extern "C" void kernel_launch(void* const* d_in, const int* in_sizes, int n_in,
                              void* d_out, int out_size, void* d_ws, size_t ws_size,
                              hipStream_t stream) {
    const float* frames = (const float*)d_in[0];
    const float* gamma  = (const float*)d_in[1];
    const float* beta   = (const float*)d_in[2];
    const float* W_in   = (const float*)d_in[3];
    const float* conv_w = (const float*)d_in[4];
    const float* conv_b = (const float*)d_in[5];
    const float* W_x    = (const float*)d_in[6];
    const float* W_dt   = (const float*)d_in[7];
    const float* b_dt   = (const float*)d_in[8];
    const float* A_log  = (const float*)d_in[9];
    const float* Dvec   = (const float*)d_in[10];
    const float* W_out  = (const float*)d_in[11];

    char* ws = (char*)d_ws;
    // workspace layout (all offsets 256B-aligned)
    _Float16* W_inT  = (_Float16*)(ws + 0);          //  2 MiB   (2048 x 512)
    _Float16* W_xT   = (_Float16*)(ws + 2097152);    //  384 KiB (192p x 1024)
    _Float16* W_dtT  = (_Float16*)(ws + 2490368);    //  64 KiB  (1024 x 32)
    _Float16* W_outT = (_Float16*)(ws + 2555904);    //  1 MiB   (512 x 1024)
    _Float16* xn16   = (_Float16*)(ws + 3604480);    //  2 MiB   (2048 x 512)
    float*    xz     = (float*)   (ws + 5701632);    // 16 MiB   (2048 x 2048)
    float*    xf32   = (float*)   (ws + 22478848);   //  8 MiB   (2048 x 1024)
    _Float16* xf16   = (_Float16*)(ws + 30867456);   //  4 MiB
    float*    xdb    = (float*)   (ws + 35061760);   //  1.5 MiB (2048 x 192p)
    _Float16* dt16   = (_Float16*)(ws + 36634624);   //  128 KiB (2048 x 32)
    float*    delta  = (float*)   (ws + 36765696);   //  8 MiB
    float*    ysb    = (float*)   (ws + 45154304);   //  8 MiB
    _Float16* y16    = (_Float16*)(ws + 53542912);   //  4 MiB

    const int EW = 256;
    // weight transpose + f16 convert
    k_transpose_cvt<<<(D_MODEL * 2 * D_INNER + EW - 1) / EW, EW, 0, stream>>>(W_in, W_inT, D_MODEL, 2 * D_INNER);
    k_transpose_cvt<<<(D_INNER * XDB_W + EW - 1) / EW, EW, 0, stream>>>(W_x, W_xT, D_INNER, XDB_W);
    k_transpose_cvt<<<(DT_RANK * D_INNER + EW - 1) / EW, EW, 0, stream>>>(W_dt, W_dtT, DT_RANK, D_INNER);
    k_transpose_cvt<<<(D_INNER * D_MODEL + EW - 1) / EW, EW, 0, stream>>>(W_out, W_outT, D_INNER, D_MODEL);

    // LayerNorm -> f16
    k_layernorm<<<NTOK, 256, 0, stream>>>(frames, gamma, beta, xn16);

    // in_proj: (2048 x 512) * (512 x 2048) -> xz
    k_wmma_gemm<0><<<dim3(2 * D_INNER / GEMM_BN, NTOK / GEMM_BM), 128, 0, stream>>>(
        xn16, W_inT, xz, NTOK, 2 * D_INNER, D_MODEL, nullptr, nullptr);

    // depthwise causal conv + SiLU
    k_conv_silu<<<(NTOK * D_INNER + EW - 1) / EW, EW, 0, stream>>>(xz, conv_w, conv_b, xf32, xf16);

    // x @ W_x: (2048 x 1024) * (1024 x 192p) -> xdb (dt | B | C | pad)
    k_wmma_gemm<0><<<dim3(XDB_P / GEMM_BN, NTOK / GEMM_BM), 128, 0, stream>>>(
        xf16, W_xT, xdb, NTOK, XDB_P, D_INNER, nullptr, nullptr);

    // delta = softplus(dt @ W_dt + b_dt): (2048 x 32) * (32 x 1024)
    k_extract_dt<<<(NTOK * DT_RANK + EW - 1) / EW, EW, 0, stream>>>(xdb, dt16);
    k_wmma_gemm<1><<<dim3(D_INNER / GEMM_BN, NTOK / GEMM_BM), 128, 0, stream>>>(
        dt16, W_dtT, delta, NTOK, D_INNER, DT_RANK, b_dt, nullptr);

    // selective scan: one wave per (b, d)
    k_scan<<<NBATCH * D_INNER, 32, 0, stream>>>(delta, xdb, xf32, A_log, ysb);

    // gate: (ys + x*D) * silu(z)
    k_gate<<<(NTOK * D_INNER + EW - 1) / EW, EW, 0, stream>>>(ysb, xf32, xz, Dvec, y16);

    // out_proj + residual: (2048 x 1024) * (1024 x 512) + frames -> d_out
    k_wmma_gemm<2><<<dim3(D_MODEL / GEMM_BN, NTOK / GEMM_BM), 128, 0, stream>>>(
        y16, W_outT, (float*)d_out, NTOK, D_MODEL, D_INNER, nullptr, frames);
}
